// Head_49074296324248
// MI455X (gfx1250) — matrix operand order repaired and checked by reading, not compiled, not run on hardware
//
#include <hip/hip_runtime.h>
#include <hip/hip_bf16.h>

// Problem constants (from reference): B=4, T=4096, D=1024, H=64
static constexpr int BB = 4;
static constexpr int TT = 4096;
static constexpr int DD = 1024;
static constexpr int HH = 64;

typedef __attribute__((ext_vector_type(16))) __bf16 v16bf;
typedef __attribute__((ext_vector_type(8)))  __bf16 v8bf;
typedef __attribute__((ext_vector_type(8)))  float  v8f;

#define WMMA_BF16(a, b, c) \
  __builtin_amdgcn_wmma_f32_16x16x32_bf16(false, (a), false, (b), (short)0, (c), false, false)

// ---------------------------------------------------------------------------
// Fragment loaders (CDNA5 ISA 7.12.2 layouts, wave32). All loads are
// contiguous 16B chunks -> global_load_b128.
// A 16x32 bf16: lane L holds row M=L%16; K = {koff..koff+7, 16+koff..} koff=8*(L/16)
// B 32x16 bf16: lane L holds col N=L%16; K = {koff..koff+7, 16+koff..} koff=8*(L/16)
// C/D 16x16 f32: lane L holds col N=L%16; VGPR r holds row M = r + 8*(L/16)
// ---------------------------------------------------------------------------

__device__ __forceinline__ v16bf load_a_bf16(const __bf16* __restrict__ src, int ld,
                                             int m0, int k0, int lane) {
  const int row  = m0 + (lane & 15);
  const int koff = (lane >> 4) * 8;
  const __bf16* p = src + (size_t)row * ld + k0 + koff;
  v8bf lo = *(const v8bf*)(p);        // K = koff..koff+7
  v8bf hi = *(const v8bf*)(p + 16);   // K = 16+koff..16+koff+7
  return __builtin_shufflevector(lo, hi, 0,1,2,3,4,5,6,7,8,9,10,11,12,13,14,15);
}

__device__ __forceinline__ v16bf load_a_f32cvt(const float* __restrict__ src, int ld,
                                               int m0, int k0, int lane) {
  const int row  = m0 + (lane & 15);
  const int koff = (lane >> 4) * 8;
  const float* p = src + (size_t)row * ld + k0 + koff;
  v8f lo = *(const v8f*)(p);
  v8f hi = *(const v8f*)(p + 16);
  v16bf a;
#pragma unroll
  for (int i = 0; i < 8; ++i) a[i]     = (__bf16)lo[i];
#pragma unroll
  for (int i = 0; i < 8; ++i) a[8 + i] = (__bf16)hi[i];
  return a;
}

// B[kk][n] = src[n0+n][k0+kk] for row-major src[N][K] (i.e. transposed operand)
__device__ __forceinline__ v16bf load_bT_bf16(const __bf16* __restrict__ src, int ld,
                                              int n0, int k0, int lane) {
  const int n  = n0 + (lane & 15);
  const int kk = k0 + (lane >> 4) * 8;
  const __bf16* p = src + (size_t)n * ld + kk;
  v8bf lo = *(const v8bf*)(p);
  v8bf hi = *(const v8bf*)(p + 16);
  return __builtin_shufflevector(lo, hi, 0,1,2,3,4,5,6,7,8,9,10,11,12,13,14,15);
}

// ---------------------------------------------------------------------------
// Kernel 0: transpose + convert the three weight matrices to bf16 W^T[H][D]
// so projection B-fragments become contiguous loads.  One-shot, 196K elems.
// ---------------------------------------------------------------------------
__global__ __launch_bounds__(256) void wtrans_kernel(const float* __restrict__ Wk,
                                                     const float* __restrict__ Wq,
                                                     const float* __restrict__ Wv,
                                                     __bf16* __restrict__ wkT,
                                                     __bf16* __restrict__ wqT,
                                                     __bf16* __restrict__ wvT) {
  const int idx = blockIdx.x * 256 + threadIdx.x;   // coalesced read side
  if (idx >= DD * HH) return;
  const int d = idx / HH;
  const int h = idx % HH;
  const size_t t = (size_t)h * DD + d;
  wkT[t] = (__bf16)Wk[idx];
  wqT[t] = (__bf16)Wq[idx];
  wvT[t] = (__bf16)Wv[idx];
}

// ---------------------------------------------------------------------------
// Kernel 1: q/k/v projections.  One wave per (b, 16-row block of T).
// q,k stored row-major bf16 [T][H]; v stored COLUMN-major bf16 [H][T] so the
// P@V WMMA B-fragments in kernel 3 are contiguous.
// ---------------------------------------------------------------------------
__global__ __launch_bounds__(32) void proj_kernel(const float* __restrict__ x,
                                                  const __bf16* __restrict__ wkT,
                                                  const __bf16* __restrict__ wqT,
                                                  const __bf16* __restrict__ wvT,
                                                  __bf16* __restrict__ qb,
                                                  __bf16* __restrict__ kb,
                                                  __bf16* __restrict__ vTb) {
  const int blk  = blockIdx.x;                 // B * T/16 blocks
  const int b    = blk / (TT / 16);
  const int i0   = (blk % (TT / 16)) * 16;
  const int lane = threadIdx.x;

  const float* xb = x + (size_t)b * TT * DD;

  v8f accK[4] = {}, accQ[4] = {}, accV[4] = {};

  for (int k0 = 0; k0 < DD; k0 += 32) {
    // prefetch next K-slab of x (gfx1250 global_prefetch_b8)
    if (k0 + 32 < DD)
      __builtin_prefetch(xb + (size_t)(i0 + (lane & 15)) * DD + k0 + 32, 0, 1);
    v16bf a = load_a_f32cvt(xb, DD, i0, k0, lane);
#pragma unroll
    for (int n = 0; n < 4; ++n) {
      v16bf bk = load_bT_bf16(wkT, DD, n * 16, k0, lane);
      accK[n] = WMMA_BF16(a, bk, accK[n]);
      v16bf bq = load_bT_bf16(wqT, DD, n * 16, k0, lane);
      accQ[n] = WMMA_BF16(a, bq, accQ[n]);
      v16bf bv = load_bT_bf16(wvT, DD, n * 16, k0, lane);
      accV[n] = WMMA_BF16(a, bv, accV[n]);
    }
  }

  const size_t base = (size_t)b * TT * HH;
  __bf16* vT = vTb + (size_t)b * HH * TT;
#pragma unroll
  for (int n = 0; n < 4; ++n) {
#pragma unroll
    for (int r = 0; r < 8; ++r) {
      const int row = i0 + r + 8 * (lane >> 4);
      const int col = n * 16 + (lane & 15);
      qb[base + (size_t)row * HH + col] = (__bf16)accQ[n][r];
      kb[base + (size_t)row * HH + col] = (__bf16)accK[n][r];
      vT[(size_t)col * TT + row]        = (__bf16)accV[n][r];
    }
  }
}

// ---------------------------------------------------------------------------
// Kernel 2: per-column softmax stats (reference softmax is over query axis i).
// One wave per (b, 16 columns j).  S tile = Q x K^T; lane L tracks 8 rows of
// column L%16 online; merged across L<->L+16 with wave32 shfl at the end.
// ---------------------------------------------------------------------------
__global__ __launch_bounds__(32) void colstats_kernel(const __bf16* __restrict__ qb,
                                                      const __bf16* __restrict__ kb,
                                                      float* __restrict__ mcol,
                                                      float* __restrict__ zcol) {
  const int blk  = blockIdx.x;                 // B * T/16 blocks
  const int b    = blk / (TT / 16);
  const int j0   = (blk % (TT / 16)) * 16;
  const int lane = threadIdx.x;

  const __bf16* q = qb + (size_t)b * TT * HH;
  const __bf16* k = kb + (size_t)b * TT * HH;

  // K^T B-fragments for this column block: fixed across the i loop.
  const v16bf b0 = load_bT_bf16(k, HH, j0, 0,  lane);
  const v16bf b1 = load_bT_bf16(k, HH, j0, 32, lane);

  float m = -1e30f, z = 0.0f;
  for (int i0 = 0; i0 < TT; i0 += 16) {
    if (i0 + 16 < TT)
      __builtin_prefetch(q + (size_t)(i0 + 16 + (lane & 15)) * HH, 0, 1);
    v16bf a0 = load_a_bf16(q, HH, i0, 0,  lane);
    v16bf a1 = load_a_bf16(q, HH, i0, 32, lane);
    v8f s = {};
    s = WMMA_BF16(a0, b0, s);
    s = WMMA_BF16(a1, b1, s);

    float tm = m;
#pragma unroll
    for (int r = 0; r < 8; ++r) tm = fmaxf(tm, s[r]);
    z *= __expf(m - tm);
#pragma unroll
    for (int r = 0; r < 8; ++r) z += __expf(s[r] - tm);
    m = tm;
  }

  // merge the two half-columns (lanes L and L^16 hold the same column j)
  const float m2 = __shfl_xor(m, 16, 32);
  const float z2 = __shfl_xor(z, 16, 32);
  const float mm = fmaxf(m, m2);
  const float zz = z * __expf(m - mm) + z2 * __expf(m2 - mm);

  if (lane < 16) {
    mcol[(size_t)b * TT + j0 + lane] = mm;
    zcol[(size_t)b * TT + j0 + lane] = zz;
  }
}

// ---------------------------------------------------------------------------
// Kernel 3: out = P @ v, P recomputed tile-by-tile (flash style), NO LDS:
// compute S^T tiles = K x Q^T.  S^T's C/D layout (lane = col i, VGPR r = row j)
// is exactly the A-fragment layout of P for the O += P[16x32] x V WMMA, so the
// two 16x16 S^T tiles fill the two halves of the P fragment in registers.
// m_j / Z_j are consumed as contiguous v8f loads (j = jt + koff + r).
// ---------------------------------------------------------------------------
__global__ __launch_bounds__(32) void attn_out_kernel(const __bf16* __restrict__ qb,
                                                      const __bf16* __restrict__ kb,
                                                      const __bf16* __restrict__ vTb,
                                                      const float* __restrict__ mcol,
                                                      const float* __restrict__ zcol,
                                                      float* __restrict__ out) {
  const int blk  = blockIdx.x;                 // B * T/16 blocks
  const int b    = blk / (TT / 16);
  const int i0   = (blk % (TT / 16)) * 16;
  const int lane = threadIdx.x;

  const __bf16* q  = qb  + (size_t)b * TT * HH;
  const __bf16* k  = kb  + (size_t)b * TT * HH;
  const __bf16* vT = vTb + (size_t)b * HH * TT;
  const float*  mc = mcol + (size_t)b * TT;
  const float*  zc = zcol + (size_t)b * TT;

  // Q^T B-fragments for this row block: fixed across the j loop.
  const v16bf bq0 = load_bT_bf16(q, HH, i0, 0,  lane);
  const v16bf bq1 = load_bT_bf16(q, HH, i0, 32, lane);
  const int koff  = (lane >> 4) * 8;

  v8f acc[4] = {};

  for (int j0 = 0; j0 < TT; j0 += 32) {
    if (j0 + 32 < TT) {
      __builtin_prefetch(k + (size_t)(j0 + 32 + (lane & 31)) * HH, 0, 1);
      __builtin_prefetch(vT + (size_t)(lane & 15) * TT + j0 + 32, 0, 1);
    }

    v16bf pa;  // P A-fragment, 16 rows i x K=32 cols j, built in registers
#pragma unroll
    for (int t = 0; t < 2; ++t) {
      const int jt = j0 + t * 16;
      v16bf ak0 = load_a_bf16(k, HH, jt, 0,  lane);
      v16bf ak1 = load_a_bf16(k, HH, jt, 32, lane);
      v8f sT = {};
      sT = WMMA_BF16(ak0, bq0, sT);    // S^T[j, i] tile
      sT = WMMA_BF16(ak1, bq1, sT);

      // lane L holds S[i0+L%16, jt+koff+r] in sT[r]; stats are contiguous in j
      const v8f mv = *(const v8f*)(mc + jt + koff);
      const v8f zv = *(const v8f*)(zc + jt + koff);
#pragma unroll
      for (int r = 0; r < 8; ++r) {
        const float pv = __expf(sT[r] - mv[r]) * __frcp_rn(zv[r]);
        pa[t * 8 + r] = (__bf16)pv;    // element r -> K=koff+r ; 8+r -> K=16+koff+r
      }
    }

    // O[16x64] += P[16x32] x V[32x64]  (V fragments contiguous from vT)
#pragma unroll
    for (int n = 0; n < 4; ++n) {
      v16bf bv = load_bT_bf16(vT, TT, n * 16, j0, lane);
      acc[n] = WMMA_BF16(pa, bv, acc[n]);
    }
  }

  float* ob = out + (size_t)b * TT * HH;
#pragma unroll
  for (int n = 0; n < 4; ++n) {
#pragma unroll
    for (int r = 0; r < 8; ++r) {
      const int row = i0 + r + 8 * (lane >> 4);
      const int col = n * 16 + (lane & 15);
      ob[(size_t)row * HH + col] = acc[n][r];
    }
  }
}

// ---------------------------------------------------------------------------
// Launch.  ws layout (all 2MB/128KB aligned blocks):
//   qb 2MB | kb 2MB | vT 2MB | wkT 128K | wqT 128K | wvT 128K | mcol 64K | zcol 64K
// ---------------------------------------------------------------------------
extern "C" void kernel_launch(void* const* d_in, const int* in_sizes, int n_in,
                              void* d_out, int out_size, void* d_ws, size_t ws_size,
                              hipStream_t stream) {
  const float* x  = (const float*)d_in[0];
  const float* Wk = (const float*)d_in[1];
  const float* Wq = (const float*)d_in[2];
  const float* Wv = (const float*)d_in[3];
  float* out = (float*)d_out;

  const size_t nQKV = (size_t)BB * TT * HH;    // 1,048,576 elems
  const size_t nW   = (size_t)DD * HH;         // 65,536 elems
  __bf16* qb  = (__bf16*)d_ws;
  __bf16* kb  = qb + nQKV;
  __bf16* vT  = kb + nQKV;
  __bf16* wkT = vT + nQKV;
  __bf16* wqT = wkT + nW;
  __bf16* wvT = wqT + nW;
  float* mcol = (float*)(wvT + nW);
  float* zcol = mcol + (size_t)BB * TT;

  const int nblk = BB * (TT / 16);             // 1024 workgroups, 1 wave each

  wtrans_kernel<<<(DD * HH + 255) / 256, 256, 0, stream>>>(Wk, Wq, Wv, wkT, wqT, wvT);
  proj_kernel<<<nblk, 32, 0, stream>>>(x, wkT, wqT, wvT, qb, kb, vT);
  colstats_kernel<<<nblk, 32, 0, stream>>>(qb, kb, mcol, zcol);
  attn_out_kernel<<<nblk, 32, 0, stream>>>(qb, kb, vT, mcol, zcol, out);
}
